// _RPN_6605659701432
// MI455X (gfx1250) — compile-verified, matrix operand order checked by reading, not run on hardware
//
#include <hip/hip_runtime.h>
#include <hip/hip_bf16.h>

// ---------------------------------------------------------------------------
// _RPN forward for MI455X (gfx1250): conv3d as 27-tap implicit GEMM on
// v_wmma_f32_16x16x32_bf16 (16x64 tile per wave, b128 fragment loads,
// zero-row pointer-select boundary masking), temporal max-pool, 1x1 heads
// as WMMA GEMMs.
// ---------------------------------------------------------------------------

typedef __attribute__((ext_vector_type(16))) __bf16 v16bf;
typedef __attribute__((ext_vector_type(8)))  float  v8f;

#define HWc   784        // 28*28
#define Tt    16
#define Cc    512
#define NPOS  25088      // 2*16*784

__device__ inline unsigned short f2bf(float f) {
    unsigned int x = __float_as_uint(f);
    x += 0x7FFFu + ((x >> 16) & 1u);          // round-to-nearest-even
    return (unsigned short)(x >> 16);
}
__device__ inline float bf2f(unsigned short u) {
    return __uint_as_float(((unsigned int)u) << 16);
}

// Fragment = 16 bf16 = two 16B chunks (dwords pbase+{0..3} and pbase+{8..11})
union Frag { uint4 q[2]; v16bf v; };

__device__ inline v8f wmma_bf16(const Frag& a, const Frag& b, v8f c) {
    return __builtin_amdgcn_wmma_f32_16x16x32_bf16(
               false, a.v, false, b.v, (short)0, c, false, false);
}

// --------------------------- pack kernels ----------------------------------

// base_feat [B,C,T,H,W] f32 -> Xbf [pos = b*T*HW + t*HW + hw][C] bf16
__global__ void k_pack_x(const float* __restrict__ X, unsigned short* __restrict__ Xbf) {
    long i = blockIdx.x * (long)blockDim.x + threadIdx.x;
    if (i >= (long)NPOS * Cc) return;
    int  c   = (int)(i & 511);
    long pos = i >> 9;
    int  b   = (int)(pos / (Tt * HWc));
    int  rem = (int)(pos % (Tt * HWc));
    int  t   = rem / HWc;
    int  hw  = rem % HWc;
    Xbf[i] = f2bf(X[(((long)(b * Cc + c) * Tt) + t) * HWc + hw]);
}

// W_conv [O,I,3,3,3] f32 -> Wc [tap][o][ci] bf16
__global__ void k_pack_wc(const float* __restrict__ W, unsigned short* __restrict__ Wc) {
    long i = blockIdx.x * (long)blockDim.x + threadIdx.x;
    if (i >= 27L * Cc * Cc) return;
    int tap = (int)(i / (Cc * Cc));
    int r   = (int)(i % (Cc * Cc));
    int o = r >> 9, ci = r & 511;
    Wc[i] = f2bf(W[((long)(o * Cc + ci)) * 27 + tap]);
}

// head weights -> single bf16 block: [cls 32(pad) | 960 | 720 | 480 | 240] x 512
__global__ void k_pack_wh(const float* __restrict__ Wcls,
                          const float* __restrict__ Wb0, const float* __restrict__ Wb1,
                          const float* __restrict__ Wb2, const float* __restrict__ Wb3,
                          unsigned short* __restrict__ Wh) {
    int i = blockIdx.x * blockDim.x + threadIdx.x;
    if (i >= 2432 * 512) return;
    int row = i >> 9, c = i & 511;
    float v;
    if      (row <   32) v = (row < 30) ? Wcls[row * 512 + c] : 0.f;
    else if (row <  992) v = Wb0[(row -   32) * 512 + c];
    else if (row < 1712) v = Wb1[(row -  992) * 512 + c];
    else if (row < 2192) v = Wb2[(row - 1712) * 512 + c];
    else                 v = Wb3[(row - 2192) * 512 + c];
    Wh[i] = f2bf(v);
}

// zero row for out-of-bounds conv taps (ws is poisoned, must clear explicitly)
__global__ void k_zero(unsigned short* __restrict__ z) {
    z[threadIdx.x] = 0;
}

// ------------------------ conv3d implicit GEMM -----------------------------
// One wave computes a 16(pos) x 64(outch) tile: 4 accumulators share one
// A fragment; K = 27 taps * 512 ci. Boundary taps read a zero row.
__global__ void k_conv(const unsigned short* __restrict__ X,    // [NPOS][512] bf16
                       const unsigned short* __restrict__ Wt,   // [27][512][512] bf16
                       const float* __restrict__ bias,
                       const unsigned short* __restrict__ zrow, // [512] zeros
                       unsigned short* __restrict__ Y) {        // [NPOS][512] bf16
    int wave = (blockIdx.x * blockDim.x + threadIdx.x) >> 5;
    int lane = threadIdx.x & 31;
    const int NG = Cc / 64;                       // 8 n-groups of 64 channels
    int mtile = wave / NG;
    int ng    = wave % NG;
    if (mtile >= NPOS / 16) return;               // wave-uniform exit

    int lr = lane & 15;
    int hi = lane >> 4;
    int m  = mtile * 16 + lr;                     // output position (row of A/D)
    int b   = m / (Tt * HWc);
    int rem = m % (Tt * HWc);
    int t   = rem / HWc;
    int hw  = rem % HWc;
    int h = hw / 28, w = hw % 28;

    const uint4* zr = (const uint4*)zrow;

    v8f acc0 = {}, acc1 = {}, acc2 = {}, acc3 = {};
    for (int tap = 0; tap < 27; ++tap) {
        int kt = tap / 9, kh = (tap / 3) % 3, kw = tap % 3;
        int ts = t + kt - 1, hs = h + kh - 1, ws = w + kw - 1;
        bool valid = (ts >= 0) & (ts < Tt) & (hs >= 0) & (hs < 28) & (ws >= 0) & (ws < 28);
        const uint4* arow = valid
            ? (const uint4*)(X + ((size_t)((b * Tt + ts) * HWc + hs * 28 + ws)) * Cc)
            : zr;                                  // unconditional b128 loads
        const unsigned short* wb = Wt + ((size_t)tap * Cc + ng * 64 + lr) * Cc;
        const uint4* b0 = (const uint4*)(wb);
        const uint4* b1 = (const uint4*)(wb + 16 * Cc);
        const uint4* b2 = (const uint4*)(wb + 32 * Cc);
        const uint4* b3 = (const uint4*)(wb + 48 * Cc);
        for (int kb = 0; kb < Cc / 32; ++kb) {
            int q0 = kb * 4 + hi;                  // 16B-chunk index (pbase = hi*4 dw)
            Frag A;  A.q[0]  = arow[q0]; A.q[1]  = arow[q0 + 2];
            Frag B;
            B.q[0] = b0[q0]; B.q[1] = b0[q0 + 2];  acc0 = wmma_bf16(A, B, acc0);
            B.q[0] = b1[q0]; B.q[1] = b1[q0 + 2];  acc1 = wmma_bf16(A, B, acc1);
            B.q[0] = b2[q0]; B.q[1] = b2[q0 + 2];  acc2 = wmma_bf16(A, B, acc2);
            B.q[0] = b3[q0]; B.q[1] = b3[q0 + 2];  acc3 = wmma_bf16(A, B, acc3);
        }
    }
    int n0 = ng * 64 + lr;
    float bv0 = bias[n0], bv1 = bias[n0 + 16], bv2 = bias[n0 + 32], bv3 = bias[n0 + 48];
    #pragma unroll
    for (int r = 0; r < 8; ++r) {
        int pos = mtile * 16 + r + hi * 8;        // D layout: VGPR r -> M=r / r+8
        size_t o = (size_t)pos * Cc + n0;
        float v0 = acc0[r] + bv0, v1 = acc1[r] + bv1;
        float v2 = acc2[r] + bv2, v3 = acc3[r] + bv3;
        Y[o]      = f2bf(v0 > 0.f ? v0 : 0.f);    // ReLU
        Y[o + 16] = f2bf(v1 > 0.f ? v1 : 0.f);
        Y[o + 32] = f2bf(v2 > 0.f ? v2 : 0.f);
        Y[o + 48] = f2bf(v3 > 0.f ? v3 : 0.f);
    }
}

// -------------------------- temporal max pool ------------------------------
__global__ void k_pool(const unsigned short* __restrict__ Y,
                       unsigned short* __restrict__ P) {
    long idx = blockIdx.x * (long)blockDim.x + threadIdx.x;
    const long c0 = 802816, c1 = 4816896, c2 = 12042240, c3 = 23281664;
    if (idx >= c3) return;
    int k, Tp; long local;
    if      (idx < c0) { k = 16; Tp = 1;  local = idx;      }
    else if (idx < c1) { k = 12; Tp = 5;  local = idx - c0; }
    else if (idx < c2) { k = 8;  Tp = 9;  local = idx - c1; }
    else               { k = 4;  Tp = 13; local = idx - c2; }
    int  c   = (int)(local & 511);
    long pos = local >> 9;
    int  b   = (int)(pos / (Tp * HWc));
    int  rem = (int)(pos % (Tp * HWc));
    int  tp  = rem / HWc, hw = rem % HWc;
    long src = ((long)(b * Tt + tp) * HWc + hw) * Cc + c;
    float m = -3.4e38f;
    for (int dt = 0; dt < k; ++dt) {
        m = fmaxf(m, bf2f(Y[src]));
        src += (long)HWc * Cc;
    }
    P[idx] = f2bf(m);
}

// ----------------------------- head GEMMs ----------------------------------
// mode 0: out[pos*Npad + n] = v                   (cls scores scratch)
// mode 1: out[((b*Npad+n)*Tp + t)*784 + hw] = v   (bbox direct to d_out, NCTHW)
__global__ void k_head(const unsigned short* __restrict__ A,   // [M][512] bf16
                       const unsigned short* __restrict__ Bw,  // [Npad][512] bf16
                       const float* __restrict__ bias, int nbias,
                       int M, int Npad,
                       float* __restrict__ out, int mode, int Tp) {
    int wave = (blockIdx.x * blockDim.x + threadIdx.x) >> 5;
    int lane = threadIdx.x & 31;
    int mt = M / 16, nt = Npad / 16;
    if (wave >= mt * nt) return;                  // wave-uniform exit
    int mtile = wave / nt, ntile = wave % nt;
    int lr = lane & 15, hi = lane >> 4;
    int n = ntile * 16 + lr;
    const uint4* arow = (const uint4*)(A  + (size_t)(mtile * 16 + lr) * Cc);
    const uint4* brow = (const uint4*)(Bw + (size_t)n * Cc);

    v8f acc = {};
    for (int kb = 0; kb < Cc / 32; ++kb) {
        int q0 = kb * 4 + hi;
        Frag Af; Af.q[0] = arow[q0]; Af.q[1] = arow[q0 + 2];
        Frag Bf; Bf.q[0] = brow[q0]; Bf.q[1] = brow[q0 + 2];
        acc = wmma_bf16(Af, Bf, acc);
    }
    float bv = (n < nbias) ? bias[n] : 0.f;
    #pragma unroll
    for (int r = 0; r < 8; ++r) {
        int pos = mtile * 16 + r + hi * 8;
        float v = acc[r] + bv;
        if (mode == 0) {
            out[(size_t)pos * Npad + n] = v;
        } else {
            int b   = pos / (Tp * HWc);
            int rem = pos % (Tp * HWc);
            int t   = rem / HWc, hw = rem % HWc;
            out[(((size_t)(b * Npad + n)) * Tp + t) * HWc + hw] = v;
        }
    }
}

// --------------------------- cls pair softmax ------------------------------
// reshape(B,2,15*T,H,W) softmax(axis=1): pair (c, c+15) -> sigmoid(s_c - s_oth)
__global__ void k_cls(const float* __restrict__ S,   // [45472][32] scores
                      float* __restrict__ out) {     // d_out cls region (offset 0)
    int idx = blockIdx.x * blockDim.x + threadIdx.x;
    const int n0 = 47040, n1 = 235200, n2 = 423360, n3 = 611520;
    if (idx >= n0 + n1 + n2 + n3) return;
    int Tp, posb, local; long ob;
    if      (idx < n0)           { Tp = 1;  posb = 0;     ob = 0;            local = idx; }
    else if (idx < n0 + n1)      { Tp = 5;  posb = 1568;  ob = n0;           local = idx - n0; }
    else if (idx < n0 + n1 + n2) { Tp = 9;  posb = 9408;  ob = n0 + n1;      local = idx - n0 - n1; }
    else                         { Tp = 13; posb = 23520; ob = n0 + n1 + n2; local = idx - n0 - n1 - n2; }
    int per_b = 30 * Tp * HWc;
    int b    = local / per_b;
    int rem  = local % per_b;
    int c    = rem / (Tp * HWc);
    int rem2 = rem % (Tp * HWc);
    int t    = rem2 / HWc, hw = rem2 % HWc;
    int pos  = posb + (b * Tp + t) * HWc + hw;
    float s_self = S[(size_t)pos * 32 + c];
    float s_oth  = S[(size_t)pos * 32 + (c < 15 ? c + 15 : c - 15)];
    out[ob + local] = 1.f / (1.f + __expf(s_oth - s_self));
}

// ------------------------------- launcher ----------------------------------

extern "C" void kernel_launch(void* const* d_in, const int* in_sizes, int n_in,
                              void* d_out, int out_size, void* d_ws, size_t ws_size,
                              hipStream_t stream) {
    const float* base_feat = (const float*)d_in[0];
    const float* W_conv    = (const float*)d_in[4];
    const float* b_conv    = (const float*)d_in[5];
    const float* W_cls     = (const float*)d_in[6];
    const float* b_cls     = (const float*)d_in[7];
    const float* W_bb [4]  = { (const float*)d_in[8],  (const float*)d_in[10],
                               (const float*)d_in[12], (const float*)d_in[14] };
    const float* b_bb [4]  = { (const float*)d_in[9],  (const float*)d_in[11],
                               (const float*)d_in[13], (const float*)d_in[15] };
    float* out = (float*)d_out;

    // ---- workspace carve-out ----
    size_t off = 0;
    auto carve = [&](size_t bytes) {
        void* p = (char*)d_ws + off;
        off += (bytes + 255) & ~(size_t)255;
        return p;
    };
    unsigned short* Xbf  = (unsigned short*)carve((size_t)NPOS * Cc * 2);       // 25.7 MB
    unsigned short* Ybf  = (unsigned short*)carve((size_t)NPOS * Cc * 2);       // 25.7 MB
    unsigned short* Wc   = (unsigned short*)carve(27L * Cc * Cc * 2);           // 14.2 MB
    unsigned short* Wh   = (unsigned short*)carve(2432L * Cc * 2);              //  2.5 MB
    unsigned short* P    = (unsigned short*)carve(23281664L * 2);               // 46.6 MB
    float*          Scls = (float*)carve(45472L * 32 * 4);                      //  5.8 MB
    unsigned short* Zrow = (unsigned short*)carve(Cc * 2);                      //  1 KB

    auto blocks = [](long n, int bs) { return (int)((n + bs - 1) / bs); };

    // 1) pack activations / weights to bf16, clear zero row
    k_pack_x <<<blocks((long)NPOS * Cc, 256), 256, 0, stream>>>(base_feat, Xbf);
    k_pack_wc<<<blocks(27L * Cc * Cc, 256), 256, 0, stream>>>(W_conv, Wc);
    k_pack_wh<<<blocks(2432L * 512, 256), 256, 0, stream>>>(W_cls, W_bb[0], W_bb[1],
                                                            W_bb[2], W_bb[3], Wh);
    k_zero   <<<1, Cc, 0, stream>>>(Zrow);

    // 2) conv3d + bias + ReLU  (1568 m-tiles * 8 n-groups waves, 8 waves/block)
    {
        long waves = (NPOS / 16) * (Cc / 64);
        k_conv<<<blocks(waves * 32, 256), 256, 0, stream>>>(Xbf, Wc, b_conv, Zrow, Ybf);
    }

    // 3) temporal max pools (all 4 branches)
    k_pool<<<blocks(23281664L, 256), 256, 0, stream>>>(Ybf, P);

    // 4) heads
    const int  Tp_[4]    = { 1, 5, 9, 13 };
    const int  Mpos[4]   = { 1568, 7840, 14112, 21952 };
    const long Pel[4]    = { 0, 802816, 4816896, 12042240 };   // P elem offsets
    const int  posB[4]   = { 0, 1568, 9408, 23520 };           // Scls pos offsets
    const int  Nbb[4]    = { 960, 720, 480, 240 };
    const long bbRow[4]  = { 32, 32 + 960, 32 + 960 + 720, 32 + 960 + 720 + 480 };
    const long bbOut[4]  = { 1317120, 2822400, 8467200, 15240960 };

    for (int br = 0; br < 4; ++br) {
        // cls head (shared weights, Npad = 32)
        {
            long waves = (long)(Mpos[br] / 16) * (32 / 16);
            k_head<<<blocks(waves * 32, 256), 256, 0, stream>>>(
                P + Pel[br], Wh /*cls rows 0..31*/, b_cls, 30,
                Mpos[br], 32, Scls + (size_t)posB[br] * 32, /*mode=*/0, Tp_[br]);
        }
        // bbox head -> directly into d_out (NCTHW)
        {
            long waves = (long)(Mpos[br] / 16) * (Nbb[br] / 16);
            k_head<<<blocks(waves * 32, 256), 256, 0, stream>>>(
                P + Pel[br], Wh + bbRow[br] * Cc, b_bb[br], Nbb[br],
                Mpos[br], Nbb[br], out + bbOut[br], /*mode=*/1, Tp_[br]);
        }
    }

    // 5) cls pairwise softmax -> d_out cls regions
    k_cls<<<blocks(1317120L, 256), 256, 0, stream>>>(Scls, out);
}
